// GQA_40750649704889
// MI455X (gfx1250) — matrix operand order repaired and checked by reading, not compiled, not run on hardware
//
#include <hip/hip_runtime.h>
#include <hip/hip_bf16.h>

// ---------------- problem constants ----------------
#define BATCH 2
#define SEQ   2048
#define DMODEL 2048
#define HQ 32
#define HKV 8
#define HD 64
#define NQKV ((HQ + 2 * HKV) * HD)   // 3072
#define GRP (HQ / HKV)               // 4

typedef __bf16 bf16;
typedef __attribute__((ext_vector_type(16))) __bf16 bf16x16;
typedef __attribute__((ext_vector_type(8)))  __bf16 bf16x8;
typedef __attribute__((ext_vector_type(8)))  float  f32x8;

// ---------------- WMMA helpers ----------------
__device__ __forceinline__ f32x8 wmma_bf16(bf16x16 a, bf16x16 b, f32x8 c) {
  // D = A(16x32 bf16) * B(32x16 bf16) + C(16x16 f32)
  return __builtin_amdgcn_wmma_f32_16x16x32_bf16(false, a, false, b, (short)0, c,
                                                 false, false);
}

// A-fragment (16x32, 16-bit): lane holds row m=lane%16.
// elements 0..7 -> K = abase+0..7, elements 8..15 -> K = abase+16..23,
// abase = 0 (lanes 0-15) or 8 (lanes 16-31).
__device__ __forceinline__ bf16x16 load_frag_a(const bf16* __restrict__ base,
                                               int row_stride, int row, int k0,
                                               int lane) {
  int abase = (lane < 16) ? 0 : 8;
  const bf16* p = base + (size_t)row * row_stride + k0 + abase;
  bf16x8 lo = *(const bf16x8*)p;
  bf16x8 hi = *(const bf16x8*)(p + 16);
  return __builtin_shufflevector(lo, hi, 0, 1, 2, 3, 4, 5, 6, 7, 8, 9, 10, 11,
                                 12, 13, 14, 15);
}

// B-fragment (32x16, 16-bit): lane holds column n=lane%16.
// elements 0..7 -> K = bbase+0..7, elements 8..15 -> K = bbase+16..23,
// bbase = 0 (lanes 0-15) or 8 (lanes 16-31).
__device__ __forceinline__ bf16x16 load_frag_b(const bf16* __restrict__ base,
                                               int row_stride, int col, int k0,
                                               int lane) {
  int bbase = (lane < 16) ? 0 : 8;
  const bf16* p = base + (size_t)col * row_stride + k0 + bbase;
  bf16x8 lo = *(const bf16x8*)p;
  bf16x8 hi = *(const bf16x8*)(p + 16);
  return __builtin_shufflevector(lo, hi, 0, 1, 2, 3, 4, 5, 6, 7, 8, 9, 10, 11,
                                 12, 13, 14, 15);
}

// ---------------- fp32 -> bf16 convert ----------------
__global__ __launch_bounds__(256) void cvt_bf16_kernel(
    const float* __restrict__ in, bf16* __restrict__ out, int n) {
  int i = (blockIdx.x * 256 + threadIdx.x) * 4;
  if (i + 3 < n) {
    float4 v = *(const float4*)(in + i);
    out[i + 0] = (bf16)v.x;
    out[i + 1] = (bf16)v.y;
    out[i + 2] = (bf16)v.z;
    out[i + 3] = (bf16)v.w;
  }
}

// ---------------- GEMM1: qkv = x @ w_qkv^T, + RoPE, scatter Q/K/Vt ----------
// A = xbf [B*L, D], B[k][n] = w_qkv[n][k]  (w row-major [NQKV, D])
// wave tile: 32 (M) x 64 (N)
__global__ __launch_bounds__(256) void gemm_qkv_rope_kernel(
    const bf16* __restrict__ A, const bf16* __restrict__ W,
    bf16* __restrict__ Q, bf16* __restrict__ K, bf16* __restrict__ Vt) {
  const int lane = threadIdx.x & 31;
  const int wave = threadIdx.x >> 5;
  const int NT = NQKV / 64;  // 48
  int wid = blockIdx.x * 8 + wave;
  int mt = wid / NT;
  int nt = wid - mt * NT;
  int m0 = mt * 32;
  int n0 = nt * 64;
  int ln = lane & 15;

  f32x8 acc[2][4] = {};
  for (int k0 = 0; k0 < DMODEL; k0 += 32) {
    bf16x16 af[2];
    af[0] = load_frag_a(A, DMODEL, m0 + ln, k0, lane);
    af[1] = load_frag_a(A, DMODEL, m0 + 16 + ln, k0, lane);
    bf16x16 bfr[4];
#pragma unroll
    for (int ni = 0; ni < 4; ni++)
      bfr[ni] = load_frag_b(W, DMODEL, n0 + ni * 16 + ln, k0, lane);
#pragma unroll
    for (int mi = 0; mi < 2; mi++)
#pragma unroll
      for (int ni = 0; ni < 4; ni++)
        acc[mi][ni] = wmma_bf16(af[mi], bfr[ni], acc[mi][ni]);
  }

  // epilogue: RoPE on q/k segments, scatter to head-major layouts
  int b = m0 / SEQ;
  int t0 = m0 - b * SEQ;
  int half8 = (lane >> 4) << 3;
#pragma unroll
  for (int mi = 0; mi < 2; mi++) {
#pragma unroll
    for (int ni = 0; ni < 4; ni++) {
      int c0 = n0 + ni * 16;
      int c = c0 + ln;
      int d = c & 63;
      if (c0 < (HQ + HKV) * HD) {  // q or k segment: apply RoPE
        // theta = ROPE_BASE^(-(d&~1)/HD); ln(10000) = 9.210340372
        float freq = __expf(-(float)(d & ~1) * (9.210340371976184f / (float)HD));
        float sgn = (d & 1) ? 1.0f : -1.0f;
#pragma unroll
        for (int j = 0; j < 8; j++) {
          float v = acc[mi][ni][j];
          float partner = __shfl_xor(v, 1, 32);
          int t = t0 + mi * 16 + j + half8;
          float ang = (float)t * freq;
          float si, co;
          __sincosf(ang, &si, &co);
          float outv = v * co + partner * si * sgn;
          if (c < HQ * HD) {
            int head = c >> 6;
            Q[(((size_t)b * HQ + head) * SEQ + t) * HD + d] = (bf16)outv;
          } else {
            int head = (c - HQ * HD) >> 6;
            K[(((size_t)b * HKV + head) * SEQ + t) * HD + d] = (bf16)outv;
          }
        }
      } else {  // v segment: store transposed [B][HKV][HD][L]
        int head = (c - (HQ + HKV) * HD) >> 6;
#pragma unroll
        for (int j = 0; j < 8; j++) {
          int t = t0 + mi * 16 + j + half8;
          Vt[(((size_t)b * HKV + head) * HD + d) * SEQ + t] =
              (bf16)acc[mi][ni][j];
        }
      }
    }
  }
}

// ---------------- flash attention (causal, GQA) ----------------
// One wave = 16 query rows. S^T = K * Q^T so softmax stats are lane-local
// and P converts in-register into the PV A-fragment. V is pre-transposed.
__global__ __launch_bounds__(256) void attn_kernel(
    const bf16* __restrict__ Q, const bf16* __restrict__ K,
    const bf16* __restrict__ Vt, bf16* __restrict__ Y) {
  const int lane = threadIdx.x & 31;
  const int wave = threadIdx.x >> 5;
  const int ln = lane & 15;
  int bh = blockIdx.y;
  int b = bh / HQ;
  int h = bh - b * HQ;
  int kvh = h / GRP;
  int q0 = blockIdx.x * 128 + wave * 16;
  int qrow = q0 + ln;

  const bf16* Qh = Q + ((size_t)b * HQ + h) * SEQ * HD;
  const bf16* Kh = K + ((size_t)b * HKV + kvh) * SEQ * HD;
  const bf16* Vh = Vt + ((size_t)b * HKV + kvh) * HD * SEQ;

  // Q^T B-fragments (two K-steps over head dim 64)
  bf16x16 qb[2];
  {
    int bbase = (lane < 16) ? 0 : 8;
    const bf16* qp = Qh + (size_t)qrow * HD + bbase;
    qb[0] = __builtin_shufflevector(*(const bf16x8*)qp, *(const bf16x8*)(qp + 16),
                                    0, 1, 2, 3, 4, 5, 6, 7, 8, 9, 10, 11, 12, 13, 14, 15);
    qb[1] = __builtin_shufflevector(*(const bf16x8*)(qp + 32), *(const bf16x8*)(qp + 48),
                                    0, 1, 2, 3, 4, 5, 6, 7, 8, 9, 10, 11, 12, 13, 14, 15);
  }

  f32x8 o[4] = {};
  float m_run = -1e30f, l_run = 0.0f;
  const float scale = 0.125f;  // 1/sqrt(64)
  int half8 = (lane >> 4) << 3;

  for (int n0 = 0; n0 <= q0 + 15; n0 += 32) {
    // S^T tiles: st0 -> keys n0..n0+15, st1 -> keys n0+16..n0+31
    f32x8 st0 = {}, st1 = {};
    {
      int abase = (lane < 16) ? 0 : 8;
      const bf16* kp0 = Kh + (size_t)(n0 + ln) * HD;
      const bf16* kp1 = kp0 + 16 * HD;
#pragma unroll
      for (int ks = 0; ks < 2; ks++) {
        bf16x8 lo0 = *(const bf16x8*)(kp0 + ks * 32 + abase);
        bf16x8 hi0 = *(const bf16x8*)(kp0 + ks * 32 + abase + 16);
        bf16x16 ka0 = __builtin_shufflevector(lo0, hi0, 0, 1, 2, 3, 4, 5, 6, 7,
                                              8, 9, 10, 11, 12, 13, 14, 15);
        st0 = wmma_bf16(ka0, qb[ks], st0);
        bf16x8 lo1 = *(const bf16x8*)(kp1 + ks * 32 + abase);
        bf16x8 hi1 = *(const bf16x8*)(kp1 + ks * 32 + abase + 16);
        bf16x16 ka1 = __builtin_shufflevector(lo1, hi1, 0, 1, 2, 3, 4, 5, 6, 7,
                                              8, 9, 10, 11, 12, 13, 14, 15);
        st1 = wmma_bf16(ka1, qb[ks], st1);
      }
    }

    // scale + causal mask + tile max
    int kb0 = n0 + half8;
    int kb1 = kb0 + 16;
    float mt = -1e30f;
#pragma unroll
    for (int j = 0; j < 8; j++) {
      float s0 = st0[j] * scale;
      float s1 = st1[j] * scale;
      if (kb0 + j > qrow) s0 = -1e30f;
      if (kb1 + j > qrow) s1 = -1e30f;
      st0[j] = s0;
      st1[j] = s1;
      mt = fmaxf(mt, fmaxf(s0, s1));
    }
    mt = fmaxf(mt, __shfl_xor(mt, 16, 32));
    float new_m = fmaxf(m_run, mt);
    float alpha = __expf(m_run - new_m);

    // P = exp(S - m); pack directly into A-fragment layout
    bf16x16 pa;
    float psum = 0.0f;
#pragma unroll
    for (int j = 0; j < 8; j++) {
      float p0 = __expf(st0[j] - new_m);
      float p1 = __expf(st1[j] - new_m);
      psum += p0 + p1;
      pa[j] = (bf16)p0;
      pa[j + 8] = (bf16)p1;
    }
    psum += __shfl_xor(psum, 16, 32);
    l_run = l_run * alpha + psum;
    m_run = new_m;

    // rescale O accumulators (rows live across VGPRs -> broadcast alpha)
#pragma unroll
    for (int j = 0; j < 8; j++) {
      float aj = __shfl(alpha, j + half8, 32);
      o[0][j] *= aj;
      o[1][j] *= aj;
      o[2][j] *= aj;
      o[3][j] *= aj;
    }

    // O += P * V  (V B-fragment in table order from transposed layout)
    int vbase = (lane < 16) ? 0 : 8;
#pragma unroll
    for (int dt = 0; dt < 4; dt++) {
      const bf16* vp = Vh + (size_t)(dt * 16 + ln) * SEQ + n0 + vbase;
      bf16x16 vb = __builtin_shufflevector(*(const bf16x8*)vp, *(const bf16x8*)(vp + 16),
                                           0, 1, 2, 3, 4, 5, 6, 7, 8, 9, 10, 11, 12, 13, 14, 15);
      o[dt] = wmma_bf16(pa, vb, o[dt]);
    }
  }

  float linv = 1.0f / l_run;
#pragma unroll
  for (int j = 0; j < 8; j++) {
    float lj = __shfl(linv, j + half8, 32);
    int t = q0 + j + half8;
    bf16* yp = Y + ((size_t)b * SEQ + t) * DMODEL + h * HD + ln;
#pragma unroll
    for (int dt = 0; dt < 4; dt++) yp[dt * 16] = (bf16)(o[dt][j] * lj);
  }
}

// ---------------- GEMM2: out = Y @ w_o^T (f32 out) ----------------
__global__ __launch_bounds__(256) void gemm_out_kernel(
    const bf16* __restrict__ A, const bf16* __restrict__ W,
    float* __restrict__ Out) {
  const int lane = threadIdx.x & 31;
  const int wave = threadIdx.x >> 5;
  const int NT = DMODEL / 64;  // 32
  int wid = blockIdx.x * 8 + wave;
  int mt = wid / NT;
  int nt = wid - mt * NT;
  int m0 = mt * 32;
  int n0 = nt * 64;
  int ln = lane & 15;

  f32x8 acc[2][4] = {};
  for (int k0 = 0; k0 < DMODEL; k0 += 32) {
    bf16x16 af[2];
    af[0] = load_frag_a(A, DMODEL, m0 + ln, k0, lane);
    af[1] = load_frag_a(A, DMODEL, m0 + 16 + ln, k0, lane);
    bf16x16 bfr[4];
#pragma unroll
    for (int ni = 0; ni < 4; ni++)
      bfr[ni] = load_frag_b(W, DMODEL, n0 + ni * 16 + ln, k0, lane);
#pragma unroll
    for (int mi = 0; mi < 2; mi++)
#pragma unroll
      for (int ni = 0; ni < 4; ni++)
        acc[mi][ni] = wmma_bf16(af[mi], bfr[ni], acc[mi][ni]);
  }

  int half8 = (lane >> 4) << 3;
#pragma unroll
  for (int mi = 0; mi < 2; mi++)
#pragma unroll
    for (int ni = 0; ni < 4; ni++) {
      int c = n0 + ni * 16 + ln;
#pragma unroll
      for (int j = 0; j < 8; j++) {
        int m = m0 + mi * 16 + j + half8;
        Out[(size_t)m * DMODEL + c] = acc[mi][ni][j];
      }
    }
}

// ---------------- host launcher ----------------
extern "C" void kernel_launch(void* const* d_in, const int* in_sizes, int n_in,
                              void* d_out, int out_size, void* d_ws,
                              size_t ws_size, hipStream_t stream) {
  const float* x = (const float*)d_in[0];       // [B, L, D]
  const float* w_qkv = (const float*)d_in[1];   // [NQKV, D]
  const float* w_o = (const float*)d_in[2];     // [D, D]
  float* out = (float*)d_out;                   // [B, L, D]

  // workspace layout (bf16), all 256B-aligned
  char* w = (char*)d_ws;
  size_t off = 0;
  bf16* xbf = (bf16*)(w + off);   off += (size_t)BATCH * SEQ * DMODEL * 2;
  bf16* wqb = (bf16*)(w + off);   off += (size_t)NQKV * DMODEL * 2;
  bf16* wob = (bf16*)(w + off);   off += (size_t)DMODEL * DMODEL * 2;
  bf16* Qb  = (bf16*)(w + off);   off += (size_t)BATCH * HQ * SEQ * HD * 2;
  bf16* Kb  = (bf16*)(w + off);   off += (size_t)BATCH * HKV * SEQ * HD * 2;
  bf16* Vtb = (bf16*)(w + off);   off += (size_t)BATCH * HKV * SEQ * HD * 2;
  bf16* Yb  = (bf16*)(w + off);   off += (size_t)BATCH * SEQ * DMODEL * 2;
  (void)ws_size;

  // 1) convert inputs to bf16
  {
    int n0 = BATCH * SEQ * DMODEL;  // 8388608
    cvt_bf16_kernel<<<n0 / 1024, 256, 0, stream>>>(x, xbf, n0);
    int n1 = NQKV * DMODEL;         // 6291456
    cvt_bf16_kernel<<<n1 / 1024, 256, 0, stream>>>(w_qkv, wqb, n1);
    int n2 = DMODEL * DMODEL;       // 4194304
    cvt_bf16_kernel<<<n2 / 1024, 256, 0, stream>>>(w_o, wob, n2);
  }

  // 2) QKV projection + RoPE + scatter (waves: 128 mtiles * 48 ntiles)
  {
    int waves = (BATCH * SEQ / 32) * (NQKV / 64);  // 6144
    gemm_qkv_rope_kernel<<<waves / 8, 256, 0, stream>>>(xbf, wqb, Qb, Kb, Vtb);
  }

  // 3) causal flash attention
  {
    dim3 grid(SEQ / 128, BATCH * HQ);  // (16, 64)
    attn_kernel<<<grid, 256, 0, stream>>>(Qb, Kb, Vtb, Yb);
  }

  // 4) output projection
  {
    int waves = (BATCH * SEQ / 32) * (DMODEL / 64);  // 4096
    gemm_out_kernel<<<waves / 8, 256, 0, stream>>>(Yb, wob, out);
  }
}
